// GAT_13967233647335
// MI455X (gfx1250) — matrix operand order repaired and checked by reading, not compiled, not run on hardware
//
#include <hip/hip_runtime.h>
#include <hip/hip_bf16.h>

typedef __attribute__((ext_vector_type(16))) __bf16 v16bf;
typedef __attribute__((ext_vector_type(8)))  __bf16 v8bf;
typedef __attribute__((ext_vector_type(4)))  __bf16 v4bf;
typedef __attribute__((ext_vector_type(8)))  float  v8f;

#define N_NODES 50000
#define N_EDGES 800000
#define ETOT    (N_EDGES + N_NODES)
#define HEADS   4
#define HIDDEN  64
#define OUT_DIM 32
#define IN_DIM  128

// ---------------------------------------------------------------------------
// monotonic float <-> uint encoding for atomicMax-based segment max
// ---------------------------------------------------------------------------
__device__ __forceinline__ unsigned flipf(float f) {
  unsigned u = __float_as_uint(f);
  unsigned mask = (unsigned)((int)u >> 31) | 0x80000000u;
  return u ^ mask;
}
__device__ __forceinline__ float unflipf(unsigned u) {
  unsigned mask = (u >> 31) ? 0x80000000u : 0xFFFFFFFFu;
  return __uint_as_float(u ^ mask);
}
#define FLIP_NEG_MAX 0x00800000u   // flip(-FLT_MAX)

// ---------------------------------------------------------------------------
// fills
// ---------------------------------------------------------------------------
__global__ void fill_f32(float* __restrict__ p, float v, int n) {
  int i = blockIdx.x * blockDim.x + threadIdx.x;
  if (i < n) p[i] = v;
}
__global__ void fill_u32(unsigned* __restrict__ p, unsigned v, int n) {
  int i = blockIdx.x * blockDim.x + threadIdx.x;
  if (i < n) p[i] = v;
}

// ---------------------------------------------------------------------------
// f32 -> bf16 row-major conversion (4 elements / thread, packed cvt)
// ---------------------------------------------------------------------------
__global__ void f32_to_bf16_v4(const float* __restrict__ in,
                               __bf16* __restrict__ out, int n4) {
  int i = blockIdx.x * blockDim.x + threadIdx.x;
  if (i >= n4) return;
  const float4 v = *(const float4*)(in + 4 * i);
  v4bf o;
  o[0] = (__bf16)v.x; o[1] = (__bf16)v.y; o[2] = (__bf16)v.z; o[3] = (__bf16)v.w;
  *(v4bf*)(out + 4 * i) = o;
}

// ---------------------------------------------------------------------------
// pack W[K x OUT] (f32, row-major) into per-wave B fragments (bf16):
// fragment (kt, ct): 32 lanes x 16 contiguous bf16; lane L (g=L>>4, n=ct*16+(L&15))
// holds K = kt*32 + 8*g + j (j<8), kt*32 + 16 + 8*g + (j-8) (j>=8)  (matches 16-bit B 32x16 WMMA layout).
// ---------------------------------------------------------------------------
template <int KT, int NCT>
__global__ void pack_B(const float* __restrict__ W, __bf16* __restrict__ Wpk,
                       int OUT) {
  int idx = blockIdx.x * blockDim.x + threadIdx.x;
  if (idx >= KT * NCT * 32) return;
  const int lane = idx & 31;
  const int t = idx >> 5;
  const int ct = t % NCT, kt = t / NCT;
  const int g = lane >> 4;
  const int n = ct * 16 + (lane & 15);
  __bf16* o = Wpk + idx * 16;
#pragma unroll
  for (int j = 0; j < 16; ++j)
    o[j] = (__bf16)W[(kt * 32 + 8 * g + j + (j >= 8 ? 8 : 0)) * OUT + n];
}

// ---------------------------------------------------------------------------
// WMMA GEMM on pre-packed operands:
//   Hout[N x OUT] = Xb[N x K_DIM](bf16,row-major) @ W(packed fragments)
// One wave per (16-row block, NT-tile column strip).  Per K-tile: A = two
// coalesced b128 loads, each B fragment = two coalesced b128 loads, NT WMMAs.
//   grid = (N/16, OUT/(16*NT)), block = 32
// ---------------------------------------------------------------------------
template <int K_DIM, int NT>
__global__ void __launch_bounds__(32)
gemm_wmma_pk(const __bf16* __restrict__ Xb, const __bf16* __restrict__ Wpk,
             float* __restrict__ Hout, int OUT) {
  const int NCT = OUT >> 4;
  const int rowBase = blockIdx.x * 16;
  const int strip = blockIdx.y;            // strip of NT column tiles
  const int lane = threadIdx.x & 31;
  const int g = lane >> 4, mr = lane & 15;

  const __bf16* arow = Xb + (rowBase + mr) * K_DIM + 8 * g;
  v8f acc[NT];
#pragma unroll
  for (int t = 0; t < NT; ++t) acc[t] = (v8f){};

#pragma unroll
  for (int kt = 0; kt < K_DIM / 32; ++kt) {
    const v8bf alo = *(const v8bf*)(arow + kt * 32);        // K = 8g .. 8g+7
    const v8bf ahi = *(const v8bf*)(arow + kt * 32 + 16);   // K = 16+8g ..
    const v16bf a = __builtin_shufflevector(alo, ahi, 0, 1, 2, 3, 4, 5, 6, 7,
                                            8, 9, 10, 11, 12, 13, 14, 15);
#pragma unroll
    for (int t = 0; t < NT; ++t) {
      const __bf16* bp = Wpk + ((kt * NCT + strip * NT + t) * 32 + lane) * 16;
      const v8bf blo = *(const v8bf*)bp;
      const v8bf bhi = *(const v8bf*)(bp + 8);
      const v16bf b = __builtin_shufflevector(blo, bhi, 0, 1, 2, 3, 4, 5, 6, 7,
                                              8, 9, 10, 11, 12, 13, 14, 15);
      acc[t] = __builtin_amdgcn_wmma_f32_16x16x32_bf16(
          false, a, false, b, (short)0, acc[t], false, false);
    }
  }
  // D layout: VGPR v holds row M = v + 8*g, col N = lane&15
#pragma unroll
  for (int t = 0; t < NT; ++t)
#pragma unroll
    for (int v = 0; v < 8; ++v)
      Hout[(rowBase + v + 8 * g) * OUT + (strip * NT + t) * 16 + mr] = acc[t][v];
}

// ---------------------------------------------------------------------------
// per-(node, head) attention coefficients:  as = <h, a_src>, ad = <h, a_dst>
// ---------------------------------------------------------------------------
template <int H, int F>
__global__ void node_alpha(const float* __restrict__ Hfeat,
                           const float* __restrict__ a_src,
                           const float* __restrict__ a_dst,
                           float* __restrict__ as, float* __restrict__ ad) {
  int idx = blockIdx.x * blockDim.x + threadIdx.x;
  if (idx >= N_NODES * H) return;
  const int h = idx % H, n = idx / H;
  const float* hp = Hfeat + (n * H + h) * F;
  float s = 0.f, d = 0.f;
#pragma unroll
  for (int f = 0; f < F; f += 4) {
    const float4 v = *(const float4*)(hp + f);
    const float4 ws = *(const float4*)(a_src + h * F + f);
    const float4 wd = *(const float4*)(a_dst + h * F + f);
    s += v.x * ws.x + v.y * ws.y + v.z * ws.z + v.w * ws.w;
    d += v.x * wd.x + v.y * wd.y + v.z * wd.z + v.w * wd.w;
  }
  as[idx] = s;
  ad[idx] = d;
}

// ---------------------------------------------------------------------------
// edge pass 1: logits = leaky_relu(as[src] + ad[dst]); segment max via
// atomicMax on monotonic uint encoding.  Edges >= N_EDGES are self-loops.
// ---------------------------------------------------------------------------
template <int H>
__global__ void edge_logits_max(const int* __restrict__ srcArr,
                                const int* __restrict__ dstArr,
                                const float* __restrict__ as,
                                const float* __restrict__ ad,
                                float* __restrict__ logits,
                                unsigned* __restrict__ smax) {
  int idx = blockIdx.x * blockDim.x + threadIdx.x;
  if (idx >= ETOT * H) return;
  const int h = idx % H, e = idx / H;
  int s, d;
  if (e < N_EDGES) { s = srcArr[e]; d = dstArr[e]; } else { s = d = e - N_EDGES; }
  float l = as[s * H + h] + ad[d * H + h];
  l = (l > 0.f) ? l : 0.2f * l;   // leaky_relu, negative_slope=0.2
  logits[idx] = l;
  atomicMax(&smax[d * H + h], flipf(l));
}

// ---------------------------------------------------------------------------
// edge pass 2: expd = exp(logit - segmax[dst]); segment sum via atomicAdd
// ---------------------------------------------------------------------------
template <int H>
__global__ void edge_exp_sum(const int* __restrict__ dstArr,
                             float* __restrict__ logits,
                             const unsigned* __restrict__ smax,
                             float* __restrict__ ssum) {
  int idx = blockIdx.x * blockDim.x + threadIdx.x;
  if (idx >= ETOT * H) return;
  const int h = idx % H, e = idx / H;
  const int d = (e < N_EDGES) ? dstArr[e] : e - N_EDGES;
  float ex = __expf(logits[idx] - unflipf(smax[d * H + h]));
  logits[idx] = ex;
  atomicAdd(&ssum[d * H + h], ex);
}

// ---------------------------------------------------------------------------
// edge pass 3: out[dst] += alpha * h[src]  (float4 chunks of feature dim)
// ---------------------------------------------------------------------------
template <int H, int F>
__global__ void edge_aggregate(const int* __restrict__ srcArr,
                               const int* __restrict__ dstArr,
                               const float* __restrict__ expd,
                               const float* __restrict__ ssum,
                               const float* __restrict__ Hfeat,
                               float* __restrict__ outAcc) {
  const int CH = F / 4;
  int idx = blockIdx.x * blockDim.x + threadIdx.x;
  if (idx >= ETOT * H * CH) return;
  const int c = idx % CH;
  const int t = idx / CH;
  const int h = t % H, e = t / H;
  int s, d;
  if (e < N_EDGES) { s = srcArr[e]; d = dstArr[e]; } else { s = d = e - N_EDGES; }
  const float alpha = expd[e * H + h] / (ssum[d * H + h] + 1e-16f);
  const float4 hv = *(const float4*)(Hfeat + (s * H + h) * F + 4 * c);
  float* po = outAcc + (d * H + h) * F + 4 * c;
  atomicAdd(po + 0, alpha * hv.x);
  atomicAdd(po + 1, alpha * hv.y);
  atomicAdd(po + 2, alpha * hv.z);
  atomicAdd(po + 3, alpha * hv.w);
}

// ---------------------------------------------------------------------------
// epilogues
// ---------------------------------------------------------------------------
__global__ void bias_elu(float* __restrict__ a, const float* __restrict__ b,
                         int n, int dim) {
  int idx = blockIdx.x * blockDim.x + threadIdx.x;
  if (idx >= n * dim) return;
  float v = a[idx] + b[idx % dim];
  a[idx] = (v > 0.f) ? v : (__expf(v) - 1.f);
}
__global__ void bias_add(float* __restrict__ a, const float* __restrict__ b,
                         int n, int dim) {
  int idx = blockIdx.x * blockDim.x + threadIdx.x;
  if (idx >= n * dim) return;
  a[idx] = a[idx] + b[idx % dim];
}

// ---------------------------------------------------------------------------
static inline unsigned cdiv(long long a, long long b) { return (unsigned)((a + b - 1) / b); }

extern "C" void kernel_launch(void* const* d_in, const int* in_sizes, int n_in,
                              void* d_out, int out_size, void* d_ws, size_t ws_size,
                              hipStream_t stream) {
  const float* x      = (const float*)d_in[0];
  const int*   edge   = (const int*)  d_in[1];   // [2, N_EDGES]
  const float* W1     = (const float*)d_in[2];
  const float* a_src1 = (const float*)d_in[3];
  const float* a_dst1 = (const float*)d_in[4];
  const float* b1     = (const float*)d_in[5];
  const float* W2     = (const float*)d_in[6];
  const float* a_src2 = (const float*)d_in[7];
  const float* a_dst2 = (const float*)d_in[8];
  const float* b2     = (const float*)d_in[9];
  float* out = (float*)d_out;                    // [N_NODES, OUT_DIM]

  const int* srcArr = edge;
  const int* dstArr = edge + N_EDGES;

  // ---- workspace layout ----
  float* ws    = (float*)d_ws;
  float* h1    = ws;                                   // N*256 f32
  float* agg1  = h1 + (size_t)N_NODES * 256;           // N*256 f32 (-> ELU'd h)
  float* lg    = agg1 + (size_t)N_NODES * 256;         // ETOT*HEADS f32
  float* as1   = lg + (size_t)ETOT * HEADS;            // N*HEADS
  float* ad1   = as1 + N_NODES * HEADS;
  unsigned* smax1 = (unsigned*)(ad1 + N_NODES * HEADS);
  float* ssum1 = (float*)(smax1 + N_NODES * HEADS);
  __bf16* Wpk1 = (__bf16*)(ssum1 + N_NODES * HEADS);   // 128*256 bf16
  __bf16* Wpk2 = Wpk1 + IN_DIM * HEADS * HIDDEN;       // 256*32  bf16
  // xb (bf16 x, N*128) aliases lg region (dead until edge phase)
  __bf16* xb = (__bf16*)lg;
  // layer-2 buffers alias the h1 region (dead after layer-1 aggregation)
  __bf16* hb = (__bf16*)h1;                            // N*256 bf16
  float* h2  = h1 + (size_t)N_NODES * 128;             // N*32 f32
  float* as2 = h2 + (size_t)N_NODES * 32;
  float* ad2 = as2 + N_NODES;
  unsigned* smax2 = (unsigned*)(ad2 + N_NODES);
  float* ssum2 = (float*)(smax2 + N_NODES);

  const int TB = 256;

  // =========================== LAYER 1 ===========================
  fill_f32<<<cdiv((long long)N_NODES * 256, TB), TB, 0, stream>>>(agg1, 0.f, N_NODES * 256);
  fill_u32<<<cdiv(N_NODES * HEADS, TB), TB, 0, stream>>>(smax1, FLIP_NEG_MAX, N_NODES * HEADS);
  fill_f32<<<cdiv(N_NODES * HEADS, TB), TB, 0, stream>>>(ssum1, 0.f, N_NODES * HEADS);

  f32_to_bf16_v4<<<cdiv((long long)N_NODES * IN_DIM / 4, TB), TB, 0, stream>>>(
      x, xb, N_NODES * IN_DIM / 4);
  pack_B<IN_DIM / 32, 16><<<cdiv((IN_DIM / 32) * 16 * 32, TB), TB, 0, stream>>>(
      W1, Wpk1, HEADS * HIDDEN);

  // h1 = x @ W1  (bf16 WMMA, 8 col tiles per wave, 32 static WMMAs)
  gemm_wmma_pk<IN_DIM, 8><<<dim3(N_NODES / 16, 2), 32, 0, stream>>>(
      xb, Wpk1, h1, HEADS * HIDDEN);

  node_alpha<HEADS, HIDDEN><<<cdiv(N_NODES * HEADS, TB), TB, 0, stream>>>(
      h1, a_src1, a_dst1, as1, ad1);

  edge_logits_max<HEADS><<<cdiv((long long)ETOT * HEADS, TB), TB, 0, stream>>>(
      srcArr, dstArr, as1, ad1, lg, smax1);
  edge_exp_sum<HEADS><<<cdiv((long long)ETOT * HEADS, TB), TB, 0, stream>>>(
      dstArr, lg, smax1, ssum1);
  edge_aggregate<HEADS, HIDDEN>
      <<<cdiv((long long)ETOT * HEADS * (HIDDEN / 4), TB), TB, 0, stream>>>(
          srcArr, dstArr, lg, ssum1, h1, agg1);

  bias_elu<<<cdiv((long long)N_NODES * 256, TB), TB, 0, stream>>>(
      agg1, b1, N_NODES, HEADS * HIDDEN);

  // =========================== LAYER 2 ===========================
  fill_f32<<<cdiv((long long)N_NODES * OUT_DIM, TB), TB, 0, stream>>>(out, 0.f, N_NODES * OUT_DIM);
  fill_u32<<<cdiv(N_NODES, TB), TB, 0, stream>>>(smax2, FLIP_NEG_MAX, N_NODES);
  fill_f32<<<cdiv(N_NODES, TB), TB, 0, stream>>>(ssum2, 0.f, N_NODES);

  f32_to_bf16_v4<<<cdiv((long long)N_NODES * 256 / 4, TB), TB, 0, stream>>>(
      agg1, hb, N_NODES * 256 / 4);
  pack_B<(HEADS * HIDDEN) / 32, 2>
      <<<cdiv(((HEADS * HIDDEN) / 32) * 2 * 32, TB), TB, 0, stream>>>(W2, Wpk2, OUT_DIM);

  // h2 = h @ W2  (bf16 WMMA, 16 static WMMAs)
  gemm_wmma_pk<HEADS * HIDDEN, 2><<<dim3(N_NODES / 16, 1), 32, 0, stream>>>(
      hb, Wpk2, h2, OUT_DIM);

  node_alpha<1, OUT_DIM><<<cdiv(N_NODES, TB), TB, 0, stream>>>(
      h2, a_src2, a_dst2, as2, ad2);

  edge_logits_max<1><<<cdiv(ETOT, TB), TB, 0, stream>>>(
      srcArr, dstArr, as2, ad2, lg, smax2);
  edge_exp_sum<1><<<cdiv(ETOT, TB), TB, 0, stream>>>(
      dstArr, lg, smax2, ssum2);
  edge_aggregate<1, OUT_DIM>
      <<<cdiv((long long)ETOT * (OUT_DIM / 4), TB), TB, 0, stream>>>(
          srcArr, dstArr, lg, ssum2, h2, out);

  bias_add<<<cdiv((long long)N_NODES * OUT_DIM, TB), TB, 0, stream>>>(
      out, b2, N_NODES, OUT_DIM);
}
